// PointnetFPModule_37684043055438
// MI455X (gfx1250) — compile-verified, matrix operand order checked by reading, not run on hardware
//
#include <hip/hip_runtime.h>

// ---- WMMA vector types (CDNA5, wave32) ----
typedef __attribute__((ext_vector_type(16))) __bf16 v16bf;
typedef __attribute__((ext_vector_type(8)))  float  v8f;
typedef __attribute__((ext_vector_type(4)))  unsigned int v4u;
typedef __attribute__((ext_vector_type(8)))  int v8i;
typedef __attribute__((ext_vector_type(4)))  int v4i;

#define BDIM 4
#define NQ   16384
#define MK   4096
#define CC1  128
#define CC2  256
#define CF   384   // CC1 + CC2
#define HCH  256

#if __has_builtin(__builtin_amdgcn_tensor_load_to_lds)
#define HAVE_TDM 1
#else
#define HAVE_TDM 0
#endif

// =============== 1) cast fp32 weights -> bf16 ===============
__global__ void cast_weights(const float* __restrict__ W1, const float* __restrict__ W2,
                             __bf16* __restrict__ W1b, __bf16* __restrict__ W2b) {
  const int n1 = HCH * CF, n2 = HCH * HCH;
  for (int i = blockIdx.x * blockDim.x + threadIdx.x; i < n1 + n2;
       i += gridDim.x * blockDim.x) {
    if (i < n1) W1b[i] = (__bf16)W1[i];
    else        W2b[i - n1] = (__bf16)W2[i - n1];
  }
}

// =============== 2) three_nn: LDS-resident brute force ===============
__launch_bounds__(256)
__global__ void three_nn_kernel(const float* __restrict__ unknown,
                                const float* __restrict__ known,
                                int*   __restrict__ idx_out,
                                float* __restrict__ w_out) {
  __shared__ float kx[MK], ky[MK], kz[MK];   // 48 KB: whole batch of known pts
  const int b = blockIdx.y;
  const float* kb = known + (size_t)b * MK * 3;
  for (int t = threadIdx.x; t < MK; t += 256) {
    kx[t] = kb[t * 3 + 0];
    ky[t] = kb[t * 3 + 1];
    kz[t] = kb[t * 3 + 2];
  }
  __syncthreads();

  const int j = blockIdx.x * 256 + threadIdx.x;
  const float* q = unknown + ((size_t)b * NQ + j) * 3;
  const float qx = q[0], qy = q[1], qz = q[2];

  float b0 = 1e30f, b1 = 1e30f, b2 = 1e30f;
  int   i0 = 0,     i1 = 0,     i2 = 0;
  for (int i = 0; i < MK; ++i) {
    const float dx = qx - kx[i], dy = qy - ky[i], dz = qz - kz[i];
    const float d = dx * dx + dy * dy + dz * dz;
    if (d < b2) {
      if (d < b1) {
        b2 = b1; i2 = i1;
        if (d < b0) { b1 = b0; i1 = i0; b0 = d; i0 = i; }
        else        { b1 = d;  i1 = i; }
      } else { b2 = d; i2 = i; }
    }
  }
  const float r0 = 1.f / (b0 + 1e-8f);
  const float r1 = 1.f / (b1 + 1e-8f);
  const float r2 = 1.f / (b2 + 1e-8f);
  const float rs = 1.f / (r0 + r1 + r2);
  const size_t o = ((size_t)b * NQ + j) * 3;
  idx_out[o + 0] = i0; idx_out[o + 1] = i1; idx_out[o + 2] = i2;
  w_out[o + 0] = r0 * rs; w_out[o + 1] = r1 * rs; w_out[o + 2] = r2 * rs;
}

// =============== 3) interpolate + concat -> bf16 feats [B][384][n] ===============
__launch_bounds__(256)
__global__ void build_feats(const float* __restrict__ known_feats,
                            const float* __restrict__ unknown_feats,
                            const int*   __restrict__ idx,
                            const float* __restrict__ w,
                            __bf16* __restrict__ feats) {
  const int c = blockIdx.x, b = blockIdx.y;
  __bf16* dst = feats + ((size_t)b * CF + c) * NQ;
  if (c < CC2) {
    const float* kf = known_feats + ((size_t)b * CC2 + c) * MK;
    for (int j = threadIdx.x; j < NQ; j += 256) {
      const size_t o = ((size_t)b * NQ + j) * 3;
      const float v = w[o + 0] * kf[idx[o + 0]] +
                      w[o + 1] * kf[idx[o + 1]] +
                      w[o + 2] * kf[idx[o + 2]];
      dst[j] = (__bf16)v;
    }
  } else {
    const float* uf = unknown_feats + ((size_t)b * CC1 + (c - CC2)) * NQ;
    for (int j = threadIdx.x; j < NQ; j += 256) dst[j] = (__bf16)uf[j];
  }
}

// =============== 4) bf16 WMMA GEMM: Y[b] = A(256xK) * Bm[b](KxNQ) ===============
// Block tile 128(M) x 64(N), K-step 32; 8 waves, each wave owns a 32x32 subtile
// as 2x2 fragments of v_wmma_f32_16x16x32_bf16.
// Double-buffered pipeline:
//   - A tile DMA'd by the Tensor Data Mover (ping-pong LDS buffers, TDM pad
//     produces the 80B padded row stride); one s_wait_tensorcnt + one barrier
//     per K-step.
//   - B tile global-loaded into registers for step i+1 BEFORE the WMMAs of
//     step i, drained to the other LDS buffer after compute.
__launch_bounds__(256)
__global__ void wmma_gemm(const __bf16* __restrict__ A,   // [256][K] row-major
                          const __bf16* __restrict__ Bm,  // [BDIM][K][NQ]
                          float* __restrict__ Y,          // [BDIM][256][NQ]
                          int K) {
  __shared__ __bf16 la[2][128 * 40];   // A tiles, row stride 40 elem (80 B)
  __shared__ __bf16 lb[2][64 * 40];    // B tiles TRANSPOSED: lb[p][col][k]

  const int tid  = threadIdx.x;
  const int lane = tid & 31;
  const int wave = tid >> 5;
  const int wm = (wave & 3) * 32;       // wave M offset in block tile
  const int wn = (wave >> 2) * 32;      // wave N offset in block tile
  const int m0 = blockIdx.y * 128;
  const int n0 = blockIdx.x * 64;
  const int b  = blockIdx.z;
  const __bf16* Bbase = Bm + (size_t)b * K * NQ;

  const int r  = lane & 15;             // row (A) / col (B) within fragment
  const int hf = lane >> 4;             // half selects K octet (ISA 7.12.2)

  const int bkrow = tid >> 3, bnc = (tid & 7) * 8;    // B tile loader mapping
  const int arow = tid >> 1, aseg = tid & 1;          // manual A loader mapping

#if HAVE_TDM
  // --- TDM descriptor for the A tile ---
  // Group0: count=1 | lds_addr | global_addr(57b) | type=2
  // Group1: data_size=2B, pad_enable, pad_interval=16DW, pad_amount=4DW,
  //         tensor_dim0=K, tensor_dim1=256, tile_dim0=32, tile_dim1=128,
  //         tensor_dim0_stride=K
  const unsigned lds_la[2] = { (unsigned)(unsigned long long)(&la[0][0]),
                               (unsigned)(unsigned long long)(&la[1][0]) };
  const unsigned g1w0 = (1u << 16) | (1u << 20) | (3u << 22) | (3u << 25);
  const v8i g1 = { (int)g1w0,
                   (int)(((unsigned)K & 0xffffu) << 16),  // tensor_dim0 lo
                   (int)((256u & 0xffffu) << 16),         // tensor_dim1 lo
                   (int)(32u << 16),                      // tile_dim0 = 32
                   128,                                   // tile_dim1 = 128
                   (int)(unsigned)K,                      // dim0 stride lo
                   0, 0 };
  const v4i gz = {};
#if __clang_major__ >= 23
  const v8i gz8 = {};
#endif
  auto issueA = [&](int kk, int p) {
    const unsigned long long ga = (unsigned long long)(A + (size_t)m0 * K + kk);
    const v4u g0 = { 1u, lds_la[p],
                     (unsigned)(ga & 0xffffffffu),
                     ((unsigned)(ga >> 32) & 0x1ffffffu) | (2u << 30) };
#if __clang_major__ >= 23
    __builtin_amdgcn_tensor_load_to_lds(g0, g1, gz, gz, gz8, 0);
#else
    __builtin_amdgcn_tensor_load_to_lds(g0, g1, gz, gz, 0);
#endif
  };
#endif

  v8f acc[2][2] = {};

  // ---- prologue: stage buffer 0 for K-step 0 ----
#if HAVE_TDM
  if (wave == 0) issueA(0, 0);
#else
  {
    const __bf16* ga = A + (size_t)(m0 + arow) * K + aseg * 16;
    *(uint4*)(&la[0][arow * 40 + aseg * 16])     = *(const uint4*)(ga);
    *(uint4*)(&la[0][arow * 40 + aseg * 16 + 8]) = *(const uint4*)(ga + 8);
  }
#endif
  {
    const uint4 v = *(const uint4*)(Bbase + (size_t)bkrow * NQ + n0 + bnc);
    const __bf16* pv = (const __bf16*)&v;
#pragma unroll
    for (int jj = 0; jj < 8; ++jj) lb[0][(bnc + jj) * 40 + bkrow] = pv[jj];
  }

  const int steps = K >> 5;
  for (int i = 0; i < steps; ++i) {
    const int p = i & 1;
#if HAVE_TDM
    if (wave == 0) __builtin_amdgcn_s_wait_tensorcnt(0);   // A tile p landed
#endif
    __syncthreads();                                        // tiles p published

    const bool more = (i + 1 < steps);
    uint4 gbv = {};
#if !HAVE_TDM
    uint4 gav0 = {}, gav1 = {};
#endif
    if (more) {
      const int kn = (i + 1) * 32;
      // launch next B tile global load (lands while WMMAs run)
      gbv = *(const uint4*)(Bbase + (size_t)(kn + bkrow) * NQ + n0 + bnc);
#if HAVE_TDM
      if (wave == 0) issueA(kn, p ^ 1);                     // async DMA of next A
#else
      const __bf16* ga = A + (size_t)(m0 + arow) * K + kn + aseg * 16;
      gav0 = *(const uint4*)(ga);
      gav1 = *(const uint4*)(ga + 8);
#endif
      if (i + 2 < steps)  // prefetch B tile i+2 (global_prefetch_b8)
        __builtin_prefetch(Bbase + (size_t)((i + 2) * 32 + bkrow) * NQ + n0 + bnc, 0, 1);
    }

    // ---- build fragments from buffer p per ISA 16-bit A/B layout ----
    union Frag { v16bf v; unsigned int u[8]; };
    Frag af[2], bfr[2];
#pragma unroll
    for (int f = 0; f < 2; ++f) {
      const int rowa = wm + f * 16 + r;
      const int colb = wn + f * 16 + r;
#pragma unroll
      for (int vv = 0; vv < 8; ++vv) {
        const int kb = (vv < 4) ? (2 * vv + 8 * hf) : (16 + 2 * (vv - 4) + 8 * hf);
        af[f].u[vv]  = *(const unsigned int*)&la[p][rowa * 40 + kb];
        bfr[f].u[vv] = *(const unsigned int*)&lb[p][colb * 40 + kb];
      }
    }
#pragma unroll
    for (int fi = 0; fi < 2; ++fi)
#pragma unroll
      for (int fj = 0; fj < 2; ++fj)
        acc[fi][fj] = __builtin_amdgcn_wmma_f32_16x16x32_bf16(
            false, af[fi].v, false, bfr[fj].v, (short)0, acc[fi][fj], false, false);

    // ---- drain staged registers into the other buffer ----
    if (more) {
      const __bf16* pv = (const __bf16*)&gbv;
#pragma unroll
      for (int jj = 0; jj < 8; ++jj) lb[p ^ 1][(bnc + jj) * 40 + bkrow] = pv[jj];
#if !HAVE_TDM
      *(uint4*)(&la[p ^ 1][arow * 40 + aseg * 16])     = gav0;
      *(uint4*)(&la[p ^ 1][arow * 40 + aseg * 16 + 8]) = gav1;
#endif
    }
  }

  // ---- epilogue: C/D layout — VGPR rr holds M = base + rr + 8*half, N = lane%16 ----
#pragma unroll
  for (int fi = 0; fi < 2; ++fi) {
    const int gm = m0 + wm + fi * 16 + hf * 8;
#pragma unroll
    for (int fj = 0; fj < 2; ++fj) {
      const int gn = n0 + wn + fj * 16 + r;
      float* yp = Y + ((size_t)b * HCH + gm) * NQ + gn;
#pragma unroll
      for (int rr = 0; rr < 8; ++rr) yp[(size_t)rr * NQ] = acc[fi][fj][rr];
    }
  }
}

// =============== 5) per-channel BN stats -> scale/shift (one block per channel) ===============
__launch_bounds__(256)
__global__ void channel_stats(const float* __restrict__ Y,
                              const float* __restrict__ gamma,
                              const float* __restrict__ beta,
                              float* __restrict__ scale, float* __restrict__ shift) {
  const int c = blockIdx.x;
  float s = 0.f, s2 = 0.f;
  for (int b = 0; b < BDIM; ++b) {
    const float* p = Y + ((size_t)b * HCH + c) * NQ;
    for (int j = threadIdx.x; j < NQ; j += 256) {
      const float v = p[j];
      s += v; s2 += v * v;
    }
  }
  __shared__ float ls[256], lq[256];
  ls[threadIdx.x] = s; lq[threadIdx.x] = s2;
  __syncthreads();
  for (int o = 128; o > 0; o >>= 1) {
    if (threadIdx.x < o) { ls[threadIdx.x] += ls[threadIdx.x + o];
                           lq[threadIdx.x] += lq[threadIdx.x + o]; }
    __syncthreads();
  }
  if (threadIdx.x == 0) {
    const float inv = 1.f / (float)(BDIM * NQ);
    const float mu  = ls[0] * inv;
    const float var = lq[0] * inv - mu * mu;
    const float sc  = gamma[c] * rsqrtf(var + 1e-5f);
    scale[c] = sc;
    shift[c] = beta[c] - mu * sc;
  }
}

// =============== 6) BN+ReLU and cast to bf16 (layer-1 activation) ===============
__launch_bounds__(256)
__global__ void bn_relu_cast(const float* __restrict__ Y, const float* __restrict__ scale,
                             const float* __restrict__ shift, __bf16* __restrict__ H) {
  const int c = blockIdx.x, b = blockIdx.y;
  const float sc = scale[c], sh = shift[c];
  const size_t base = ((size_t)b * HCH + c) * NQ;
  for (int j = threadIdx.x; j < NQ; j += 256) {
    const float v = fmaf(Y[base + j], sc, sh);
    H[base + j] = (__bf16)(v > 0.f ? v : 0.f);
  }
}

// =============== 7) BN+ReLU in place (final output) ===============
__launch_bounds__(256)
__global__ void bn_relu_inplace(float* __restrict__ Y, const float* __restrict__ scale,
                                const float* __restrict__ shift) {
  const int c = blockIdx.x, b = blockIdx.y;
  const float sc = scale[c], sh = shift[c];
  const size_t base = ((size_t)b * HCH + c) * NQ;
  for (int j = threadIdx.x; j < NQ; j += 256) {
    const float v = fmaf(Y[base + j], sc, sh);
    Y[base + j] = v > 0.f ? v : 0.f;
  }
}

extern "C" void kernel_launch(void* const* d_in, const int* in_sizes, int n_in,
                              void* d_out, int out_size, void* d_ws, size_t ws_size,
                              hipStream_t stream) {
  (void)in_sizes; (void)n_in; (void)out_size; (void)ws_size;
  const float* unknown       = (const float*)d_in[0];
  const float* known         = (const float*)d_in[1];
  const float* unknown_feats = (const float*)d_in[2];
  const float* known_feats   = (const float*)d_in[3];
  const float* W1     = (const float*)d_in[4];
  const float* gamma1 = (const float*)d_in[5];
  const float* beta1  = (const float*)d_in[6];
  const float* W2     = (const float*)d_in[7];
  const float* gamma2 = (const float*)d_in[8];
  const float* beta2  = (const float*)d_in[9];
  float* out = (float*)d_out;

  // ---- workspace carve-up (all offsets 256B aligned) ----
  char* ws = (char*)d_ws;
  size_t off = 0;
  auto carve = [&](size_t bytes) {
    void* p = ws + off;
    off += (bytes + 255) & ~(size_t)255;
    return p;
  };
  int*    idx_buf = (int*)   carve((size_t)BDIM * NQ * 3 * sizeof(int));
  float*  w_buf   = (float*) carve((size_t)BDIM * NQ * 3 * sizeof(float));
  __bf16* W1b     = (__bf16*)carve((size_t)HCH * CF  * sizeof(__bf16));
  __bf16* W2b     = (__bf16*)carve((size_t)HCH * HCH * sizeof(__bf16));
  __bf16* feats   = (__bf16*)carve((size_t)BDIM * CF * NQ * sizeof(__bf16));
  float*  y1      = (float*) carve((size_t)BDIM * HCH * NQ * sizeof(float));
  float*  scale1  = (float*) carve(HCH * sizeof(float));
  float*  shift1  = (float*) carve(HCH * sizeof(float));
  float*  scale2  = (float*) carve(HCH * sizeof(float));
  float*  shift2  = (float*) carve(HCH * sizeof(float));
  __bf16* h       = feats;  // alias: feats dead after GEMM1, h (33MB) fits in feats (50MB)

  // 1) weights -> bf16
  cast_weights<<<160, 256, 0, stream>>>(W1, W2, W1b, W2b);
  // 2) three_nn (known pts LDS-resident per batch)
  three_nn_kernel<<<dim3(NQ / 256, BDIM), 256, 0, stream>>>(unknown, known, idx_buf, w_buf);
  // 3) interpolate + concat -> bf16 feats
  build_feats<<<dim3(CF, BDIM), 256, 0, stream>>>(known_feats, unknown_feats,
                                                  idx_buf, w_buf, feats);
  // 4) GEMM1: y1 = W1 * feats  (K=384)
  wmma_gemm<<<dim3(NQ / 64, HCH / 128, BDIM), 256, 0, stream>>>(W1b, feats, y1, CF);
  // 5) BN1 stats
  channel_stats<<<HCH, 256, 0, stream>>>(y1, gamma1, beta1, scale1, shift1);
  // 6) h = bf16(relu(bn1(y1)))
  bn_relu_cast<<<dim3(HCH, BDIM), 256, 0, stream>>>(y1, scale1, shift1, h);
  // 7) GEMM2: out = W2 * h  (K=256), pre-BN values into d_out
  wmma_gemm<<<dim3(NQ / 64, HCH / 128, BDIM), 256, 0, stream>>>(W2b, h, out, HCH);
  // 8) BN2 stats
  channel_stats<<<HCH, 256, 0, stream>>>(out, gamma2, beta2, scale2, shift2);
  // 9) out = relu(bn2(out)) in place
  bn_relu_inplace<<<dim3(HCH, BDIM), 256, 0, stream>>>(out, scale2, shift2);
}
